// MultiModalCrossAttention_35442070126893
// MI455X (gfx1250) — compile-verified, matrix operand order checked by reading, not run on hardware
//
#include <hip/hip_runtime.h>

// ---------------------------------------------------------------------------
// MultiModalCrossAttention for MI455X (gfx1250, wave32, WMMA)
//   B=4, ST=SI=2048, DIM=512
//   bf16 WMMA (V_WMMA_F32_16X16X32_BF16) for all GEMMs, f32 accumulate.
// ---------------------------------------------------------------------------

typedef __attribute__((ext_vector_type(16))) __bf16 v16bf;
typedef __attribute__((ext_vector_type(8)))  float  v8f;

#define B_   4
#define ST_  2048
#define SI_  2048
#define DIM_ 512

__device__ __forceinline__ unsigned short f2bf(float f) {
  union { float f; unsigned u; } x; x.f = f;
  unsigned r = x.u + 0x7FFFu + ((x.u >> 16) & 1u);   // round-to-nearest-even
  return (unsigned short)(r >> 16);
}

// ---------------------------------------------------------------------------
// Kernel 1: f32 -> bf16 convert, 8 elements per thread iteration (b128 I/O)
// ---------------------------------------------------------------------------
__global__ void cvt_f32_bf16(const float* __restrict__ src,
                             unsigned short* __restrict__ dst, int n8) {
  int i = blockIdx.x * blockDim.x + threadIdx.x;
  int stride = gridDim.x * blockDim.x;
  for (; i < n8; i += stride) {
    const float4* s = reinterpret_cast<const float4*>(src) + 2 * (size_t)i;
    float4 a = s[0], b = s[1];
    uint4 o;
    o.x = (unsigned)f2bf(a.x) | ((unsigned)f2bf(a.y) << 16);
    o.y = (unsigned)f2bf(a.z) | ((unsigned)f2bf(a.w) << 16);
    o.z = (unsigned)f2bf(b.x) | ((unsigned)f2bf(b.y) << 16);
    o.w = (unsigned)f2bf(b.z) | ((unsigned)f2bf(b.w) << 16);
    reinterpret_cast<uint4*>(dst)[i] = o;
  }
}

// ---------------------------------------------------------------------------
// Kernel 2: projection GEMM  Y[b,s,e] = sum_d X[b,s,d] * W[e,d] + bias[e]
//   Block: 256 threads (8 waves). Block tile = 16 (s-rows) x 512 (all e).
//   Wave w owns e-columns [64w, 64w+64) as four 16x16 WMMA accumulators.
//   X tile (16x512 bf16, 16KB) staged once in LDS and shared by all waves.
//   If transposed: write Y^T [B, 512, S]  (used for V so PV is contiguous).
// ---------------------------------------------------------------------------
__global__ __launch_bounds__(256) void proj_wmma(
    const unsigned short* __restrict__ X,   // [B,S,512] bf16
    const unsigned short* __restrict__ W,   // [512,512] bf16 row-major (e,d)
    const float* __restrict__ bias,         // [512] f32
    unsigned short* __restrict__ Y,         // bf16 out
    int S, int transposed) {
  __shared__ __align__(16) unsigned short xS[16 * DIM_];   // 16 KB

  const int tid  = threadIdx.x;
  const int wave = tid >> 5;
  const int lane = tid & 31;
  const int b     = blockIdx.y;
  const int sbase = blockIdx.x * 16;

  // stage X tile: 16*512 bf16 = 1024 x uint4
  {
    const uint4* src = reinterpret_cast<const uint4*>(X + ((size_t)b * S + sbase) * DIM_);
    uint4* dst = reinterpret_cast<uint4*>(xS);
#pragma unroll
    for (int k = 0; k < 4; ++k) dst[tid + k * 256] = src[tid + k * 256];
  }
  __syncthreads();

  v8f acc[4];
#pragma unroll
  for (int t = 0; t < 4; ++t)
#pragma unroll
    for (int r = 0; r < 8; ++r) acc[t][r] = 0.0f;

  const int c0   = (lane & 16) ? 8 : 0;   // half-wave K-chunk offset
  const int arow = lane & 15;
  const int nb   = wave * 64;

  for (int kk = 0; kk < 16; ++kk) {       // K = 512 in steps of 32
    v16bf afrag;
    {
      uint4* ap = reinterpret_cast<uint4*>(&afrag);
      const unsigned short* aptr = xS + arow * DIM_ + kk * 32 + c0;
      ap[0] = *reinterpret_cast<const uint4*>(aptr);
      ap[1] = *reinterpret_cast<const uint4*>(aptr + 16);
    }
#pragma unroll
    for (int t = 0; t < 4; ++t) {
      const int e = nb + t * 16 + (lane & 15);
      v16bf bfrag;
      uint4* bp = reinterpret_cast<uint4*>(&bfrag);
      const unsigned short* bptr = W + (size_t)e * DIM_ + kk * 32 + c0;
      bp[0] = *reinterpret_cast<const uint4*>(bptr);
      bp[1] = *reinterpret_cast<const uint4*>(bptr + 16);
      acc[t] = __builtin_amdgcn_wmma_f32_16x16x32_bf16(
          false, afrag, false, bfrag, (short)0, acc[t], false, false);
    }
  }

  const int rowoff = (lane & 16) ? 8 : 0;
#pragma unroll
  for (int t = 0; t < 4; ++t) {
    const int e = nb + t * 16 + (lane & 15);
    const float bv = bias[e];
    if (!transposed) {
#pragma unroll
      for (int r = 0; r < 8; ++r) {
        const int srow = sbase + r + rowoff;
        Y[((size_t)b * S + srow) * DIM_ + e] = f2bf(acc[t][r] + bv);
      }
    } else {
#pragma unroll
      for (int r = 0; r < 8; ++r) {
        const int srow = sbase + r + rowoff;
        Y[((size_t)b * DIM_ + e) * (size_t)S + srow] = f2bf(acc[t][r] + bv);
      }
    }
  }
}

// ---------------------------------------------------------------------------
// Kernel 3: fused flash-style cross attention
//   O[b,q,:] = softmax(Q K^T / sqrt(512)) V
//   Block: 256 threads (8 waves), one 16-query tile, key tile = 128.
//   Wave w computes score columns [16w,16w+16) using TWO independent WMMA
//   accumulator chains (even/odd K-steps) for matrix-pipe ILP; for PV it
//   owns d-slice [64w, 64w+64) (four independent 16x16 f32 accumulators).
//   V supplied transposed so PV B-fragments are contiguous b128 loads.
// ---------------------------------------------------------------------------
__global__ __launch_bounds__(256) void flash_attn_wmma(
    const unsigned short* __restrict__ Q,   // [B,Sq,512] bf16
    const unsigned short* __restrict__ K,   // [B,Sk,512] bf16
    const unsigned short* __restrict__ VT,  // [B,512,Sk] bf16 (transposed V)
    float* __restrict__ O,                  // [B,Sq,512] f32
    int Sq, int Sk) {
  __shared__ __align__(16) unsigned short qS[16 * DIM_];  // 16 KB
  __shared__ __align__(16) float          Sb[16 * 128];   //  8 KB scores
  __shared__ __align__(16) unsigned short Pb[16 * 128];   //  4 KB probs bf16
  __shared__ float mrow[16], lrow[16], crow[16];

  const int tid  = threadIdx.x;
  const int wave = tid >> 5;
  const int lane = tid & 31;
  const int b     = blockIdx.y;
  const int sbase = blockIdx.x * 16;
  const float scale = 0.04419417382415922f;   // 1/sqrt(512)

  // stage q tile
  {
    const uint4* src = reinterpret_cast<const uint4*>(Q + ((size_t)b * Sq + sbase) * DIM_);
    uint4* dst = reinterpret_cast<uint4*>(qS);
#pragma unroll
    for (int k2 = 0; k2 < 4; ++k2) dst[tid + k2 * 256] = src[tid + k2 * 256];
  }
  if (tid < 16) { mrow[tid] = -1e30f; lrow[tid] = 0.0f; }
  __syncthreads();

  v8f acc[4];
#pragma unroll
  for (int t = 0; t < 4; ++t)
#pragma unroll
    for (int r = 0; r < 8; ++r) acc[t][r] = 0.0f;

  const int c0     = (lane & 16) ? 8 : 0;
  const int arow   = lane & 15;
  const int rowoff = (lane & 16) ? 8 : 0;
  const unsigned short* Kb = K  + (size_t)b * Sk * DIM_;
  const unsigned short* Vb = VT + (size_t)b * DIM_ * Sk;

  for (int j = 0; j < Sk; j += 128) {
    // ---- scores: this wave's 16 keys, two independent WMMA chains -------
    v8f sacc0, sacc1;
#pragma unroll
    for (int r = 0; r < 8; ++r) { sacc0[r] = 0.0f; sacc1[r] = 0.0f; }

    const int key = j + wave * 16 + (lane & 15);
    const unsigned short* kcol = Kb + (size_t)key * DIM_;
    if (j + 128 < Sk)
      __builtin_prefetch(Kb + (size_t)(key + 128) * DIM_, 0, 1);

#pragma unroll
    for (int kk = 0; kk < 16; kk += 2) {
      v16bf a0, a1, b0, b1;
      {
        uint4* ap = reinterpret_cast<uint4*>(&a0);
        const unsigned short* aptr = qS + arow * DIM_ + kk * 32 + c0;
        ap[0] = *reinterpret_cast<const uint4*>(aptr);
        ap[1] = *reinterpret_cast<const uint4*>(aptr + 16);
        uint4* ap1 = reinterpret_cast<uint4*>(&a1);
        ap1[0] = *reinterpret_cast<const uint4*>(aptr + 32);
        ap1[1] = *reinterpret_cast<const uint4*>(aptr + 48);
      }
      {
        uint4* bp = reinterpret_cast<uint4*>(&b0);
        const unsigned short* bptr = kcol + kk * 32 + c0;
        bp[0] = *reinterpret_cast<const uint4*>(bptr);
        bp[1] = *reinterpret_cast<const uint4*>(bptr + 16);
        uint4* bp1 = reinterpret_cast<uint4*>(&b1);
        bp1[0] = *reinterpret_cast<const uint4*>(bptr + 32);
        bp1[1] = *reinterpret_cast<const uint4*>(bptr + 48);
      }
      sacc0 = __builtin_amdgcn_wmma_f32_16x16x32_bf16(
          false, a0, false, b0, (short)0, sacc0, false, false);
      sacc1 = __builtin_amdgcn_wmma_f32_16x16x32_bf16(
          false, a1, false, b1, (short)0, sacc1, false, false);
    }
#pragma unroll
    for (int r = 0; r < 8; ++r)
      Sb[(r + rowoff) * 128 + wave * 16 + (lane & 15)] =
          (sacc0[r] + sacc1[r]) * scale;
    __syncthreads();

    // ---- online softmax over the 16x128 score block ---------------------
    {
      const int row = tid >> 4, sub = tid & 15;  // 16 threads/row (one half-wave)
      float v[8]; float mloc = -1e30f;
#pragma unroll
      for (int i = 0; i < 8; ++i) {
        v[i] = Sb[row * 128 + sub * 8 + i];
        mloc = fmaxf(mloc, v[i]);
      }
#pragma unroll
      for (int off = 8; off >= 1; off >>= 1)
        mloc = fmaxf(mloc, __shfl_xor(mloc, off, 16));
      const float mold = mrow[row];
      const float mnew = fmaxf(mold, mloc);
      const float c    = __expf(mold - mnew);
      float ls = 0.0f;
#pragma unroll
      for (int i = 0; i < 8; ++i) {
        const float p = __expf(v[i] - mnew);
        ls += p;
        Pb[row * 128 + sub * 8 + i] = f2bf(p);
      }
#pragma unroll
      for (int off = 8; off >= 1; off >>= 1)
        ls += __shfl_xor(ls, off, 16);
      if (sub == 0) {
        mrow[row] = mnew;
        crow[row] = c;
        lrow[row] = lrow[row] * c + ls;
      }
    }
    __syncthreads();

    // ---- rescale old accumulators, then O += P @ V ----------------------
    float cf[8];
#pragma unroll
    for (int r = 0; r < 8; ++r) cf[r] = crow[r + rowoff];
#pragma unroll
    for (int t = 0; t < 4; ++t)
#pragma unroll
      for (int r = 0; r < 8; ++r) acc[t][r] *= cf[r];

#pragma unroll
    for (int kk = 0; kk < 4; ++kk) {       // 128 keys = 4 x K32
      v16bf pfrag;
      {
        uint4* ap = reinterpret_cast<uint4*>(&pfrag);
        const unsigned short* aptr = Pb + arow * 128 + kk * 32 + c0;
        ap[0] = *reinterpret_cast<const uint4*>(aptr);
        ap[1] = *reinterpret_cast<const uint4*>(aptr + 16);
      }
#pragma unroll
      for (int t = 0; t < 4; ++t) {
        const int d = wave * 64 + t * 16 + (lane & 15);
        v16bf vfrag;
        uint4* bp = reinterpret_cast<uint4*>(&vfrag);
        const unsigned short* bptr = Vb + (size_t)d * Sk + j + kk * 32 + c0;
        bp[0] = *reinterpret_cast<const uint4*>(bptr);
        bp[1] = *reinterpret_cast<const uint4*>(bptr + 16);
        acc[t] = __builtin_amdgcn_wmma_f32_16x16x32_bf16(
            false, pfrag, false, vfrag, (short)0, acc[t], false, false);
      }
    }
  }

  // ---- epilogue: normalize by row sums, write f32 ------------------------
  float lv[8];
#pragma unroll
  for (int r = 0; r < 8; ++r) lv[r] = 1.0f / lrow[r + rowoff];
#pragma unroll
  for (int t = 0; t < 4; ++t) {
    const int d = wave * 64 + t * 16 + (lane & 15);
#pragma unroll
    for (int r = 0; r < 8; ++r) {
      const int srow = sbase + r + rowoff;
      O[((size_t)b * Sq + srow) * DIM_ + d] = acc[t][r] * lv[r];
    }
  }
}

// ---------------------------------------------------------------------------
// Launch
// ---------------------------------------------------------------------------
extern "C" void kernel_launch(void* const* d_in, const int* in_sizes, int n_in,
                              void* d_out, int out_size, void* d_ws, size_t ws_size,
                              hipStream_t stream) {
  (void)in_sizes; (void)n_in; (void)out_size; (void)ws_size;

  const size_t SX = (size_t)B_ * ST_ * DIM_;   // 4,194,304 elems per tensor
  const size_t SW = (size_t)DIM_ * DIM_;       //   262,144 elems per weight

  unsigned short* ws = (unsigned short*)d_ws;
  // bf16 activation copies
  unsigned short* bx[6];
  for (int i = 0; i < 6; ++i) bx[i] = ws + i * SX;
  // bf16 weight copies
  unsigned short* bw[6];
  unsigned short* wbase = ws + 6 * SX;
  for (int i = 0; i < 6; ++i) bw[i] = wbase + i * SW;
  // projected tensors (bf16); V projections stored transposed [B,512,S]
  unsigned short* pr   = wbase + 6 * SW;
  unsigned short* q_t  = pr + 0 * SX;
  unsigned short* k_t  = pr + 1 * SX;
  unsigned short* vT_t = pr + 2 * SX;
  unsigned short* q_i  = pr + 3 * SX;
  unsigned short* k_i  = pr + 4 * SX;
  unsigned short* vT_i = pr + 5 * SX;

  // 1) converts: activations (d_in[0..5]) and weights (d_in[6,8,10,12,14,16])
  for (int i = 0; i < 6; ++i)
    cvt_f32_bf16<<<1024, 256, 0, stream>>>((const float*)d_in[i], bx[i], (int)(SX / 8));
  const int widx[6] = {6, 8, 10, 12, 14, 16};
  for (int i = 0; i < 6; ++i)
    cvt_f32_bf16<<<64, 256, 0, stream>>>((const float*)d_in[widx[i]], bw[i], (int)(SW / 8));

  // 2) projections  (bias pointers are the f32 inputs directly)
  dim3 pg(ST_ / 16, B_);
  proj_wmma<<<pg, 256, 0, stream>>>(bx[0], bw[0], (const float*)d_in[7],  q_t,  ST_, 0);
  proj_wmma<<<pg, 256, 0, stream>>>(bx[1], bw[1], (const float*)d_in[9],  k_t,  ST_, 0);
  proj_wmma<<<pg, 256, 0, stream>>>(bx[2], bw[2], (const float*)d_in[11], vT_t, ST_, 1);
  proj_wmma<<<pg, 256, 0, stream>>>(bx[3], bw[3], (const float*)d_in[13], q_i,  SI_, 0);
  proj_wmma<<<pg, 256, 0, stream>>>(bx[4], bw[4], (const float*)d_in[15], k_i,  SI_, 0);
  proj_wmma<<<pg, 256, 0, stream>>>(bx[5], bw[5], (const float*)d_in[17], vT_i, SI_, 1);

  // 3) fused attention: h_text = attn(q_t, k_i, v_i); h_image = attn(q_i, k_t, v_t)
  float* h_text  = (float*)d_out;
  float* h_image = h_text + SX;
  dim3 ag(ST_ / 16, B_);
  flash_attn_wmma<<<ag, 256, 0, stream>>>(q_t, k_i, vT_i, h_text,  ST_, SI_);
  flash_attn_wmma<<<ag, 256, 0, stream>>>(q_i, k_t, vT_t, h_image, SI_, ST_);
}
